// SelfAttention_61795989455368
// MI455X (gfx1250) — matrix operand order repaired and checked by reading, not compiled, not run on hardware
//
#include <hip/hip_runtime.h>

// ---------------------------------------------------------------------------
// Self-attention (B=8, C=512, CQK=64, N=2048), fp32 in/out, bf16 WMMA compute.
// Pipeline:
//   1) swizzle x / W into WMMA-fragment-ready bf16 layouts (global, L2 resident)
//   2) WMMA projections -> q,k,v (fp32 plain)
//   3) swizzle q,k,v into A/B fragment layouts
//   4) fused flash-attention: QK^T -> online softmax -> P*V, all WMMA bf16
// ---------------------------------------------------------------------------

typedef __bf16 v16bf __attribute__((ext_vector_type(16)));
typedef float  v8f   __attribute__((ext_vector_type(8)));

#define BATCH 8
#define CDIM  512
#define CQK   64
#define NDIM  2048

__device__ __forceinline__ unsigned short f32_to_bf16(float f) {
  unsigned u = __float_as_uint(f);
  u += 0x7fffu + ((u >> 16) & 1u);   // round to nearest even
  return (unsigned short)(u >> 16);
}
__device__ __forceinline__ unsigned pack_bf16(float lo, float hi) {
  return (unsigned)f32_to_bf16(lo) | ((unsigned)f32_to_bf16(hi) << 16);
}

// Load one 512-element bf16 fragment (A or B operand): lane reads 32B.
__device__ __forceinline__ v16bf load_frag(const unsigned short* frag, int lane) {
  union { uint4 q[2]; v16bf v; } u;
  const uint4* p = (const uint4*)frag;
  u.q[0] = p[lane * 2 + 0];
  u.q[1] = p[lane * 2 + 1];
  return u.v;
}

__device__ __forceinline__ v8f wmma_bf16(v16bf a, v16bf b, v8f c) {
  return __builtin_amdgcn_wmma_f32_16x16x32_bf16(
      /*neg_a=*/false, a, /*neg_b=*/false, b,
      /*c_mod=*/(short)0, c, /*reuse_a=*/false, /*reuse_b=*/false);
}

// A-layout K index for (vgpr v, lane-half hiL): contiguous order k = 16*hiL + 2*v
__device__ __forceinline__ int a_klocal(int v, int hiL) {
  return 2 * v + 16 * hiL;  // pair base (h=0)
}

// --------------------------- swizzle kernels -------------------------------
// XB[b][nt16][kt32=16][512]: B-fragments of X (32c x 16n tiles), bf16
__global__ void swizzle_x_kernel(const float* __restrict__ x, unsigned* __restrict__ XB) {
  int d = blockIdx.x * blockDim.x + threadIdx.x;     // 2^22 dwords
  int p  = d & 255;
  int kt = (d >> 8) & 15;
  int nt = (d >> 12) & 127;
  int b  = d >> 19;
  int L = p >> 3, v = p & 7;
  int c = kt * 32 + 2 * v + 16 * (L >> 4);
  int n = nt * 16 + (L & 15);
  const float* xb = x + (size_t)b * CDIM * NDIM;
  XB[d] = pack_bf16(xb[(size_t)c * NDIM + n], xb[(size_t)(c + 1) * NDIM + n]);
}

// WA[mt][kt=16][512]: A-fragments of a weight matrix W[M][512]
__global__ void swizzle_w_kernel(const float* __restrict__ W, unsigned* __restrict__ WA) {
  int d = blockIdx.x * blockDim.x + threadIdx.x;     // MT*16*256 dwords
  int p  = d & 255;
  int kt = (d >> 8) & 15;
  int mt = d >> 12;
  int L = p >> 3, v = p & 7, hiL = L >> 4;
  int mrow = mt * 16 + (L & 15);
  int K0 = kt * 32 + a_klocal(v, hiL);
  const float* wr = W + (size_t)mrow * CDIM;
  WA[d] = pack_bf16(wr[K0], wr[K0 + 1]);
}

// QA[b][it128][kt=2][512]: A-fragments of Q^T (16i x 32c), from qf[b][64][2048]
__global__ void swizzle_q_kernel(const float* __restrict__ qf, unsigned* __restrict__ QA) {
  int d = blockIdx.x * blockDim.x + threadIdx.x;     // 2^19 dwords
  int p  = d & 255;
  int kt = (d >> 8) & 1;
  int it = (d >> 9) & 127;
  int b  = d >> 16;
  int L = p >> 3, v = p & 7, hiL = L >> 4;
  int i  = it * 16 + (L & 15);
  int K0 = kt * 32 + a_klocal(v, hiL);
  const float* q0 = qf + (size_t)(b * CQK + K0) * NDIM + i;
  QA[d] = pack_bf16(q0[0], q0[NDIM]);
}

// KB[b][jt128][kt=2][512]: B-fragments of K (32c x 16j), from kf[b][64][2048]
__global__ void swizzle_k_kernel(const float* __restrict__ kf, unsigned* __restrict__ KB) {
  int d = blockIdx.x * blockDim.x + threadIdx.x;     // 2^19 dwords
  int p  = d & 255;
  int kt = (d >> 8) & 1;
  int jt = (d >> 9) & 127;
  int b  = d >> 16;
  int L = p >> 3, v = p & 7, hiL = L >> 4;
  int j  = jt * 16 + (L & 15);
  int K0 = kt * 32 + 2 * v + 16 * hiL;
  const float* k0 = kf + (size_t)(b * CQK + K0) * NDIM + j;
  KB[d] = pack_bf16(k0[0], k0[NDIM]);
}

// VB[b][jt64][ct32][512]: B-fragments of V^T (32j x 16c), from vf[b][512][2048]
__global__ void swizzle_v_kernel(const float* __restrict__ vf, unsigned* __restrict__ VB) {
  int d = blockIdx.x * blockDim.x + threadIdx.x;     // 2^22 dwords
  int p  = d & 255;
  int ct = (d >> 8) & 31;
  int jt = (d >> 13) & 63;
  int b  = d >> 19;
  int L = p >> 3, v = p & 7, hiL = L >> 4;
  int c  = ct * 16 + (L & 15);
  int j0 = jt * 32 + 2 * v + 16 * hiL;
  const float* vp = vf + (size_t)(b * CDIM + c) * NDIM + j0;
  VB[d] = pack_bf16(vp[0], vp[1]);
}

// ----------------------------- projection ----------------------------------
// One wave computes one 16x16 output tile: out[b][m][n] = W x + bias
__global__ __launch_bounds__(256) void proj_kernel(
    const unsigned short* __restrict__ WA, const unsigned short* __restrict__ XB,
    const float* __restrict__ bias, float* __restrict__ out, int MT, int Mrows) {
  int lane = threadIdx.x & 31;
  int wid  = threadIdx.x >> 5;
  int t = blockIdx.x * 8 + wid;
  int total = BATCH * MT * 128;
  if (t >= total) return;                 // wave-uniform
  int nt = t % 128;
  int mt = (t / 128) % MT;
  int b  = t / (128 * MT);
  v8f acc = {};
#pragma unroll
  for (int kt = 0; kt < 16; ++kt) {
    v16bf a  = load_frag(WA + ((size_t)mt * 16 + kt) * 512, lane);
    v16bf bx = load_frag(XB + (((size_t)(b * 128 + nt)) * 16 + kt) * 512, lane);
    acc = wmma_bf16(a, bx, acc);
  }
  int hiL = lane >> 4, col = lane & 15;
  int n = nt * 16 + col;
#pragma unroll
  for (int r = 0; r < 8; ++r) {
    int m = mt * 16 + r + 8 * hiL;
    out[(size_t)(b * Mrows + m) * NDIM + n] = acc[r] + bias[m];
  }
}

// --------------------------- flash attention -------------------------------
__device__ __forceinline__ float rowreduce_max(float x) {
  x = fmaxf(x, __shfl_xor(x, 1, 32));
  x = fmaxf(x, __shfl_xor(x, 2, 32));
  x = fmaxf(x, __shfl_xor(x, 4, 32));
  x = fmaxf(x, __shfl_xor(x, 8, 32));
  return x;
}
__device__ __forceinline__ float rowreduce_sum(float x) {
  x += __shfl_xor(x, 1, 32);
  x += __shfl_xor(x, 2, 32);
  x += __shfl_xor(x, 4, 32);
  x += __shfl_xor(x, 8, 32);
  return x;
}

// Block = 8 waves. Wave w handles i-tile (blockIdx.x*4 + w/2) and channel half (w&1).
__global__ __launch_bounds__(256) void attn_kernel(
    const unsigned short* __restrict__ QA, const unsigned short* __restrict__ KB,
    const unsigned short* __restrict__ VB, const float* __restrict__ x,
    const float* __restrict__ gamma_p, float* __restrict__ out) {
  __shared__ __align__(16) unsigned short pbuf[8][16][32];  // per-wave P tile (16i x 32j bf16)

  const int lane = threadIdx.x & 31;
  const int w    = threadIdx.x >> 5;
  const int hiL  = lane >> 4, col = lane & 15;
  const int b    = blockIdx.y;
  const int it   = blockIdx.x * 4 + (w >> 1);
  const int chalf = w & 1;               // 0 -> channels 0..255, 1 -> 256..511

  // Persistent Q^T A-fragments (K = 0..31, 32..63)
  const v16bf qa0 = load_frag(QA + ((size_t)(b * 128 + it) * 2 + 0) * 512, lane);
  const v16bf qa1 = load_frag(QA + ((size_t)(b * 128 + it) * 2 + 1) * 512, lane);

  v8f acc[16];
  float m[8], l[8];
#pragma unroll
  for (int t = 0; t < 16; ++t) acc[t] = v8f{};
#pragma unroll
  for (int r = 0; r < 8; ++r) { m[r] = -1e30f; l[r] = 0.f; }

  for (int jt = 0; jt < 64; ++jt) {       // 32 keys per step
    const unsigned short* kb = KB + (size_t)(b * 128 + 2 * jt) * 2 * 512;
    v8f s0 = {}, s1 = {};
    s0 = wmma_bf16(qa0, load_frag(kb + 0 * 512, lane), s0);
    s0 = wmma_bf16(qa1, load_frag(kb + 1 * 512, lane), s0);
    s1 = wmma_bf16(qa0, load_frag(kb + 2 * 512, lane), s1);
    s1 = wmma_bf16(qa1, load_frag(kb + 3 * 512, lane), s1);

    // Online softmax over the 32 new columns; C/D row r lives in lanes of one VGPR.
#pragma unroll
    for (int r = 0; r < 8; ++r) {
      float mx = rowreduce_max(fmaxf(s0[r], s1[r]));
      float nm = fmaxf(m[r], mx);
      float sc = __expf(m[r] - nm);
      m[r] = nm;
      float p0 = __expf(s0[r] - nm);
      float p1 = __expf(s1[r] - nm);
      pbuf[w][r + 8 * hiL][col]      = f32_to_bf16(p0);
      pbuf[w][r + 8 * hiL][col + 16] = f32_to_bf16(p1);
      l[r] = l[r] * sc + rowreduce_sum(p0 + p1);
#pragma unroll
      for (int t = 0; t < 16; ++t) acc[t][r] *= sc;
    }

    // Transpose P (C/D layout -> A layout) via LDS; same-wave DS ops are in-order.
    asm volatile("s_wait_dscnt 0" ::: "memory");
    v16bf pa;
    {
      union { uint4 q[2]; v16bf v; } u;
      const uint4* rp = (const uint4*)(&pbuf[w][col][0]);  // one 64B row
      u.q[0] = rp[2 * hiL];       // cols 0-7 / 16-23
      u.q[1] = rp[2 * hiL + 1];   // cols 8-15 / 24-31
      pa = u.v;
    }

    // O += P * V^T over this wave's 256 channels (16 WMMAs)
    const unsigned short* vb = VB + ((size_t)(b * 64 + jt) * 32 + chalf * 16) * 512;
#pragma unroll
    for (int t = 0; t < 16; ++t)
      acc[t] = wmma_bf16(pa, load_frag(vb + (size_t)t * 512, lane), acc[t]);
  }

  // Epilogue: normalize, gamma * out + x
  const float g = gamma_p[0];
#pragma unroll
  for (int r = 0; r < 8; ++r) {
    const float inv = 1.0f / l[r];
    const int i = it * 16 + r + 8 * hiL;
#pragma unroll
    for (int t = 0; t < 16; ++t) {
      const int c = chalf * 256 + t * 16 + col;
      const size_t idx = (size_t)(b * CDIM + c) * NDIM + i;
      out[idx] = g * (acc[t][r] * inv) + x[idx];
    }
  }
}

// ------------------------------- launch ------------------------------------
extern "C" void kernel_launch(void* const* d_in, const int* in_sizes, int n_in,
                              void* d_out, int out_size, void* d_ws, size_t ws_size,
                              hipStream_t stream) {
  (void)in_sizes; (void)n_in; (void)out_size; (void)ws_size;
  const float* x     = (const float*)d_in[0];
  const float* Wq    = (const float*)d_in[1];
  const float* bq    = (const float*)d_in[2];
  const float* Wk    = (const float*)d_in[3];
  const float* bk    = (const float*)d_in[4];
  const float* Wv    = (const float*)d_in[5];
  const float* bv    = (const float*)d_in[6];
  const float* gamma = (const float*)d_in[7];
  float* out = (float*)d_out;

  char* w = (char*)d_ws;
  size_t off = 0;
  unsigned short* XB  = (unsigned short*)(w + off); off += 16777216;  // X B-frags (reused for VB)
  unsigned short* WAq = (unsigned short*)(w + off); off += 65536;
  unsigned short* WAk = (unsigned short*)(w + off); off += 65536;
  unsigned short* WAv = (unsigned short*)(w + off); off += 524288;
  float* qf = (float*)(w + off); off += 4194304;                       // [8][64][2048] f32
  float* kf = (float*)(w + off); off += 4194304;
  float* vf = (float*)(w + off); off += 33554432;                      // [8][512][2048] f32
  unsigned short* QA = (unsigned short*)(w + off); off += 2097152;
  unsigned short* KB = (unsigned short*)(w + off); off += 2097152;
  unsigned short* VB = XB;  // XB is dead after projections; alias to save 16 MB

  // 1) swizzle inputs into fragment layouts
  swizzle_x_kernel<<<16384, 256, 0, stream>>>(x, (unsigned*)XB);
  swizzle_w_kernel<<<64,  256, 0, stream>>>(Wq, (unsigned*)WAq);
  swizzle_w_kernel<<<64,  256, 0, stream>>>(Wk, (unsigned*)WAk);
  swizzle_w_kernel<<<512, 256, 0, stream>>>(Wv, (unsigned*)WAv);

  // 2) projections (WMMA): q,k (64 rows), v (512 rows)
  proj_kernel<<<512,  256, 0, stream>>>(WAq, XB, bq, qf, 4,  CQK);
  proj_kernel<<<512,  256, 0, stream>>>(WAk, XB, bk, kf, 4,  CQK);
  proj_kernel<<<4096, 256, 0, stream>>>(WAv, XB, bv, vf, 32, CDIM);

  // 3) swizzle q,k,v into attention fragment layouts
  swizzle_q_kernel<<<2048,  256, 0, stream>>>(qf, (unsigned*)QA);
  swizzle_k_kernel<<<2048,  256, 0, stream>>>(kf, (unsigned*)KB);
  swizzle_v_kernel<<<16384, 256, 0, stream>>>(vf, (unsigned*)VB);

  // 4) fused flash attention + residual epilogue
  dim3 grid(32, BATCH);
  attn_kernel<<<grid, 256, 0, stream>>>(QA, KB, VB, x, gamma, out);
}
